// PureGNN_20272245637431
// MI455X (gfx1250) — matrix operand order repaired and checked by reading, not compiled, not run on hardware
//
#include <hip/hip_runtime.h>

#define NNODES 100000
#define NEDGES 1600000
#define IN_DIM 4
#define HID 64
#define NLAYERS 3

typedef __attribute__((ext_vector_type(16))) __bf16 v16bf;
typedef __attribute__((ext_vector_type(8)))  __bf16 v8bf;
typedef __attribute__((ext_vector_type(8)))  float  v8f;

__device__ __forceinline__ __bf16 to_bf16(float x) { return (__bf16)x; }

// Native V_TANH_F32 on gfx1250 (single TRANS32 op, co-issues with WMMA);
// falls back to OCML tanhf if the builtin isn't declared.
#if __has_builtin(__builtin_amdgcn_tanhf)
__device__ __forceinline__ float fast_tanh(float x) { return __builtin_amdgcn_tanhf(x); }
#else
__device__ __forceinline__ float fast_tanh(float x) { return tanhf(x); }
#endif

// ---------------------------------------------------------------------------
// Kernel 1: h = tanh(X @ W_in + b_in)   [N, HID]   (tiny: 51 MFLOP)
// ---------------------------------------------------------------------------
__global__ void gnn_input_mlp(const float* __restrict__ X,
                              const float* __restrict__ Win,
                              const float* __restrict__ bin,
                              float* __restrict__ h)
{
    int idx = blockIdx.x * blockDim.x + threadIdx.x;
    if (idx >= NNODES * HID) return;
    int node = idx >> 6;
    int col  = idx & 63;
    float acc = bin[col];
#pragma unroll
    for (int i = 0; i < IN_DIM; ++i)
        acc += X[node * IN_DIM + i] * Win[i * HID + col];
    h[idx] = fast_tanh(acc);
}

// ---------------------------------------------------------------------------
// Kernel 1b: bf16 shadow of h for the WMMA gather path (h read ~32x/layer,
// written once -> convert once, halve gather bytes, kill in-loop cvt ops).
// ---------------------------------------------------------------------------
__global__ void gnn_h_to_bf16(const float* __restrict__ h,
                              __bf16* __restrict__ hbf)
{
    int i = blockIdx.x * blockDim.x + threadIdx.x;
    if (i < NNODES * HID) hbf[i] = to_bf16(h[i]);
}

// ---------------------------------------------------------------------------
// Kernel 2 (per layer): edge GEMM with bf16 WMMA + tanh + scatter-add.
// One wave handles a tile of 16 edges:
//   A  = [hbf[src] | hbf[dst]] : 16 x 128 (bf16), gathered per-lane (b128 loads)
//   B  = Ws[l]                 : 128 x 64 (bf16), register-resident fragments
//   D  = tanh(A@B + b)         : 16 x 64 (f32), scattered via atomicAdd
// 16 v_wmma_f32_16x16x32_bf16 per tile (4 K-steps x 4 N-tiles).
// ---------------------------------------------------------------------------
__global__ void __launch_bounds__(256) gnn_edge_layer(
    const __bf16* __restrict__ hbf,
    float* __restrict__ h_next,
    const int* __restrict__ src,
    const int* __restrict__ dst,
    const float* __restrict__ W,     // [2*HID, HID] row-major, f32
    const float* __restrict__ bias,  // [HID]
    int n_edges)
{
    const int lane = threadIdx.x & 31;
    const int m    = lane & 15;   // row of A-tile / column of D-tile
    const int hi   = lane >> 4;
    const int wave = blockIdx.x * (blockDim.x >> 5) + (threadIdx.x >> 5);
    const int n_waves = gridDim.x * (blockDim.x >> 5);

    // ---- B fragments: loaded once per wave (L2-hit), held in VGPRs for the
    //      whole edge loop. Lane holds col = nt*16+m; lanes 0-15 K=0..7,16..23,
    //      lanes 16-31 K=8..15,24..31 of each 32-deep K-step.
    v16bf Bfrag[4][4];
#pragma unroll
    for (int kk = 0; kk < 4; ++kk) {
#pragma unroll
        for (int nt = 0; nt < 4; ++nt) {
            const int col = nt * 16 + m;
#pragma unroll
            for (int v = 0; v < 8; ++v) {
                const int k = kk * 32 + 2 * v + 8 * hi + ((v >> 2) << 3);
                Bfrag[kk][nt][2 * v]     = to_bf16(W[k * HID + col]);
                Bfrag[kk][nt][2 * v + 1] = to_bf16(W[(k + 1) * HID + col]);
            }
        }
    }
    float bn[4];
#pragma unroll
    for (int nt = 0; nt < 4; ++nt) bn[nt] = bias[nt * 16 + m];

    const int n_tiles = n_edges >> 4;  // NEDGES % 16 == 0
    for (int t = wave; t < n_tiles; t += n_waves) {
        const int base = t << 4;
        const int e  = base + m;
        const int si = src[e];
        const int di = dst[e];
        const __bf16* __restrict__ srow = hbf + (long)si * HID;
        const __bf16* __restrict__ drow = hbf + (long)di * HID;

        v8f acc[4] = {v8f{}, v8f{}, v8f{}, v8f{}};

#pragma unroll
        for (int kk = 0; kk < 4; ++kk) {
            // Concat columns: K-steps 0,1 read h[src], 2,3 read h[dst].
            const __bf16* rowp = (kk < 2) ? srow : drow;
            const int cb = (kk & 1) * 32;
            // 16-bit A 16x32 layout: lanes 0-15: VGPR0-3 K=0..7, VGPR4-7 K=16..23
            //                        lanes 16-31: +8 on each K.
            // Two contiguous 8x bf16 (16 B) chunks per lane -> b128 loads.
            const v8bf alo = *(const v8bf*)(rowp + cb + 8 * hi);
            const v8bf ahi = *(const v8bf*)(rowp + cb + 16 + 8 * hi);
            const v16bf A = __builtin_shufflevector(
                alo, ahi, 0, 1, 2, 3, 4, 5, 6, 7, 8, 9, 10, 11, 12, 13, 14, 15);
#pragma unroll
            for (int nt = 0; nt < 4; ++nt) {
                acc[nt] = __builtin_amdgcn_wmma_f32_16x16x32_bf16(
                    /*neg_a=*/false, A, /*neg_b=*/false, Bfrag[kk][nt],
                    /*c_mod=*/(short)0, acc[nt],
                    /*reuse_a=*/false, /*reuse_b=*/false);
            }
        }

        // Epilogue: D[m_row][nt*16+m] -> atomicAdd(h_next[dst[m_row]][col]).
        // D layout: VGPR r holds row r (lanes 0-15) / row r+8 (lanes 16-31).
#pragma unroll
        for (int r = 0; r < 8; ++r) {
            const int row   = r + 8 * hi;
            const int dnode = __shfl(di, row, 32);  // dst idx lives in lane 'row'
            float* outrow = h_next + (long)dnode * HID;
#pragma unroll
            for (int nt = 0; nt < 4; ++nt) {
                atomicAdd(outrow + nt * 16 + m, fast_tanh(acc[nt][r] + bn[nt]));
            }
        }
    }
}

// ---------------------------------------------------------------------------
// Kernel 3: out = tanh(h @ W_out1 + b1) @ W_out2 + b2    [N, 3]
// One wave per node; lane j owns hidden units j and j+32; __shfl_xor reduce.
// ---------------------------------------------------------------------------
__global__ void __launch_bounds__(256) gnn_output_mlp(
    const float* __restrict__ h,
    const float* __restrict__ W1, const float* __restrict__ b1,
    const float* __restrict__ W2, const float* __restrict__ b2,
    float* __restrict__ out)
{
    const int lane = threadIdx.x & 31;
    const int node = blockIdx.x * (blockDim.x >> 5) + (threadIdx.x >> 5);
    if (node >= NNODES) return;
    const float* hrow = h + (long)node * HID;

    float a0 = b1[lane], a1 = b1[lane + 32];
    for (int i = 0; i < HID; ++i) {
        const float hv = hrow[i];
        a0 += hv * W1[i * HID + lane];
        a1 += hv * W1[i * HID + lane + 32];
    }
    const float t0 = fast_tanh(a0), t1 = fast_tanh(a1);

    float p[3];
#pragma unroll
    for (int c = 0; c < 3; ++c)
        p[c] = t0 * W2[lane * 3 + c] + t1 * W2[(lane + 32) * 3 + c];
#pragma unroll
    for (int off = 16; off > 0; off >>= 1) {
        p[0] += __shfl_xor(p[0], off, 32);
        p[1] += __shfl_xor(p[1], off, 32);
        p[2] += __shfl_xor(p[2], off, 32);
    }
    if (lane == 0) {
        out[node * 3 + 0] = p[0] + b2[0];
        out[node * 3 + 1] = p[1] + b2[1];
        out[node * 3 + 2] = p[2] + b2[2];
    }
}

// ---------------------------------------------------------------------------
extern "C" void kernel_launch(void* const* d_in, const int* in_sizes, int n_in,
                              void* d_out, int out_size, void* d_ws, size_t ws_size,
                              hipStream_t stream)
{
    const float* X    = (const float*)d_in[0];
    const int*   eidx = (const int*)d_in[1];    // [2, NEDGES]
    const float* Win  = (const float*)d_in[2];
    const float* bin  = (const float*)d_in[3];
    const float* Ws   = (const float*)d_in[4];  // [3, 128, 64]
    const float* bs   = (const float*)d_in[5];  // [3, 64]
    const float* W1   = (const float*)d_in[6];
    const float* b1o  = (const float*)d_in[7];
    const float* W2   = (const float*)d_in[8];
    const float* b2o  = (const float*)d_in[9];

    const int* src = eidx;
    const int* dst = eidx + NEDGES;

    // Scratch layout: f32 ping-pong h buffers (2 x 25.6 MB) + bf16 shadow.
    const size_t hbytes = (size_t)NNODES * HID * sizeof(float);
    float*  hA  = (float*)d_ws;
    float*  hB  = (float*)((char*)d_ws + hbytes);
    __bf16* hbf = (__bf16*)((char*)d_ws + 2 * hbytes);

    const int nelem = NNODES * HID;
    gnn_input_mlp<<<(nelem + 255) / 256, 256, 0, stream>>>(X, Win, bin, hA);

    float* hcur = hA;
    float* hnext = hB;
    for (int l = 0; l < NLAYERS; ++l) {
        // bf16 shadow of current h for the gather/WMMA path.
        gnn_h_to_bf16<<<(nelem + 255) / 256, 256, 0, stream>>>(hcur, hbf);
        // h_next starts as a copy of h; scatter-add accumulates on top.
        hipMemcpyAsync(hnext, hcur, hbytes, hipMemcpyDeviceToDevice, stream);
        gnn_edge_layer<<<2048, 256, 0, stream>>>(
            hbf, hnext, src, dst,
            Ws + (size_t)l * 2 * HID * HID, bs + (size_t)l * HID, NEDGES);
        float* tmp = hcur; hcur = hnext; hnext = tmp;
    }

    gnn_output_mlp<<<(NNODES + 7) / 8, 256, 0, stream>>>(
        hcur, W1, b1o, W2, b2o, (float*)d_out);
}
